// EMGMambaRegressor_52089363366467
// MI455X (gfx1250) — matrix operand order repaired and checked by reading, not compiled, not run on hardware
//
#include <hip/hip_runtime.h>
#include <hip/hip_bf16.h>

// ---------------------------------------------------------------------------
// EMG Mamba regressor for MI455X (gfx1250, wave32, WMMA, async LDS DMA).
// B=16, T=2048, C=12, H=256, DEPTH=5, OUT=10, DS=16, DC=4, DI=512, DR=16
// ---------------------------------------------------------------------------

typedef __attribute__((ext_vector_type(16))) _Float16 v16h;
typedef __attribute__((ext_vector_type(8)))  float    v8f;

#define BB   16
#define TT   2048
#define CC   12
#define HH   256
#define NDEPTH 5
#define NOUT 10
#define NDS  16
#define NDC  4
#define NDI  512
#define NDR  16
#define MT   ((size_t)BB * TT)   // 32768 rows

// ------------------------------ front conv ---------------------------------
__global__ void front_conv_kernel(const float* __restrict__ x,
                                  const float* __restrict__ w3, const float* __restrict__ b3,
                                  const float* __restrict__ w5, const float* __restrict__ b5,
                                  const float* __restrict__ w7, const float* __restrict__ b7,
                                  float* __restrict__ h) {
  size_t idx = blockIdx.x * (size_t)blockDim.x + threadIdx.x;   // B*T*H
  if (idx >= MT * HH) return;
  int hc = (int)(idx % HH);
  size_t bt = idx / HH;
  int t = (int)(bt % TT);
  int b = (int)(bt / TT);
  float acc = b3[hc] + b5[hc] + b7[hc];
  for (int c = 0; c < CC; ++c) {
    const float* xb = x + ((size_t)b * TT) * CC + c;
    float s = 0.f;
    for (int j = 0; j < 3; ++j) { int tt = t + j - 1; if (tt >= 0 && tt < TT) s += xb[(size_t)tt * CC] * w3[hc*36 + c*3 + j]; }
    for (int j = 0; j < 5; ++j) { int tt = t + j - 2; if (tt >= 0 && tt < TT) s += xb[(size_t)tt * CC] * w5[hc*60 + c*5 + j]; }
    for (int j = 0; j < 7; ++j) { int tt = t + j - 3; if (tt >= 0 && tt < TT) s += xb[(size_t)tt * CC] * w7[hc*84 + c*7 + j]; }
    acc += s;
  }
  h[idx] = acc * (1.f / 3.f);
}

// -------------------------- f32 -> f16 conversions -------------------------
__global__ void cvt_f16_kernel(const float* __restrict__ src, _Float16* __restrict__ dst, size_t n) {
  size_t i = blockIdx.x * (size_t)blockDim.x + threadIdx.x;
  if (i < n) dst[i] = (_Float16)src[i];
}

// dst (Kout x N, row-major f16) = transpose of src (N x Kin, row-major f32), zero pad rows >= Kin
__global__ void cvt_transpose_pad_kernel(const float* __restrict__ src, _Float16* __restrict__ dst,
                                         int N, int Kin, int Kout) {
  size_t i = blockIdx.x * (size_t)blockDim.x + threadIdx.x;
  size_t total = (size_t)Kout * N;
  if (i >= total) return;
  int n = (int)(i % N);
  int k = (int)(i / N);
  dst[i] = (k < Kin) ? (_Float16)src[(size_t)n * Kin + k] : (_Float16)0.0f;
}

// dtr extraction: A (MT x 32 f16), cols 0..15 from dbl (MT x 48), rest zero
__global__ void pack_dtr_kernel(const float* __restrict__ dbl, _Float16* __restrict__ dtA) {
  size_t i = blockIdx.x * (size_t)blockDim.x + threadIdx.x;   // MT*32
  if (i >= MT * 32) return;
  int k = (int)(i & 31);
  size_t row = i >> 5;
  dtA[i] = (k < NDR) ? (_Float16)dbl[row * 48 + k] : (_Float16)0.0f;
}

// ------------------------------ WMMA GEMM ----------------------------------
// C (MxN f32) = A (MxK f16, row-major) * B (KxN f16, row-major)
// Block: 256 threads (8 waves). Tile: 128(M) x 64(N). K step 32.
// A tile staged via GLOBAL_LOAD_ASYNC_TO_LDS_B128 (ASYNCcnt DMA path, no VGPR
// round trip); B tile staged via VGPRs because it is transposed in flight.
// NGUARD=true only for the N=48 (x_proj) instance, keeping other loops clean.
#define LDS_STR 40   // padded row stride in halves (80 bytes -> conflict-free b128 reads)

__device__ __forceinline__ void async_copy_b128x2(uint32_t lds_byte_addr, uint64_t gaddr) {
  // INST_OFFSET is added to BOTH the LDS and global addresses (ISA 10.2),
  // so one base pair covers a 32-byte row chunk with two b128 ops.
  asm volatile(
      "global_load_async_to_lds_b128 %0, %1, off\n\t"
      "global_load_async_to_lds_b128 %0, %1, off offset:16"
      :: "v"(lds_byte_addr), "v"(gaddr) : "memory");
}

__device__ __forceinline__ void wait_asynccnt0() {
  asm volatile("s_wait_asynccnt 0x0" ::: "memory");
}

template <bool NGUARD>
__global__ __launch_bounds__(256) void wmma_gemm_kernel(const _Float16* __restrict__ A,
                                                        const _Float16* __restrict__ B,
                                                        float* __restrict__ C,
                                                        int M, int N, int K) {
  __shared__ __align__(16) _Float16 As[128 * LDS_STR];
  __shared__ __align__(16) _Float16 Bs[64  * LDS_STR];
  const int m0 = blockIdx.x * 128;
  const int n0 = blockIdx.y * 64;
  const int tid  = threadIdx.x;
  const int wave = tid >> 5, lane = tid & 31;
  const int wm = wave & 3, wn = wave >> 2;
  const int lrow = lane & 15, lhalf = lane >> 4;

  // per-thread staging coordinates
  const int ar = tid >> 1, ac = tid & 1;           // A: row 0..127, 32B chunk 0..1
  const uint32_t a_lds = (uint32_t)(size_t)(As + ar * LDS_STR + ac * 16);
  const _Float16* a_g  = A + (size_t)(m0 + ar) * K + ac * 16;
  const int bk = tid >> 3, bn = tid & 7;           // B: k row 0..31, 8-col chunk 0..7
  const int nbase = n0 + bn * 8;

  v8f acc[2][2] = {};

  for (int k0 = 0; k0 < K; k0 += 32) {
    // --- A tile (128 x 32 halves): async DMA global -> LDS ---
    async_copy_b128x2(a_lds, (uint64_t)(size_t)(a_g + k0));
    if (k0 + 32 < K) __builtin_prefetch(a_g + k0 + 32, 0, 0);

    // --- B tile (32 x 64) transposed into Bs[n][k] via VGPRs ---
    {
      __align__(16) _Float16 tmp[8];
      if (!NGUARD || (nbase + 8 <= N)) {
        *(uint4*)tmp = *(const uint4*)(B + (size_t)(k0 + bk) * N + nbase);
      } else {
        for (int j = 0; j < 8; ++j)
          tmp[j] = (nbase + j < N) ? B[(size_t)(k0 + bk) * N + nbase + j] : (_Float16)0.0f;
      }
      for (int j = 0; j < 8; ++j)
        Bs[(bn * 8 + j) * LDS_STR + bk] = tmp[j];
    }
    wait_asynccnt0();          // our wave's async LDS writes are done
    __syncthreads();           // + everyone's ds stores

    // --- fragments per ISA 7.12.2 layouts ---
    v16h afrag[2], bfrag[2];
#pragma unroll
    for (int mt = 0; mt < 2; ++mt) {
      int mr = wm * 32 + mt * 16 + lrow;
      union { uint4 u[2]; v16h v; } u;
      // K runs [half*8 .. +7] and [16+half*8 .. +7] : two contiguous 16B reads
      u.u[0] = *(const uint4*)(As + mr * LDS_STR + lhalf * 8);
      u.u[1] = *(const uint4*)(As + mr * LDS_STR + 16 + lhalf * 8);
      afrag[mt] = u.v;
    }
#pragma unroll
    for (int nt = 0; nt < 2; ++nt) {
      int nr = wn * 32 + nt * 16 + lrow;
      union { uint4 u[2]; v16h v; } u;
      // lane holds column nr, K runs [half*8 .. +7] and [16+half*8 .. +7]
      u.u[0] = *(const uint4*)(Bs + nr * LDS_STR + lhalf * 8);
      u.u[1] = *(const uint4*)(Bs + nr * LDS_STR + 16 + lhalf * 8);
      bfrag[nt] = u.v;
    }
#pragma unroll
    for (int mt = 0; mt < 2; ++mt)
#pragma unroll
      for (int nt = 0; nt < 2; ++nt)
        acc[mt][nt] = __builtin_amdgcn_wmma_f32_16x16x32_f16(
            false, afrag[mt], false, bfrag[nt], (short)0, acc[mt][nt], false, false);
    __syncthreads();
  }

  // --- store D: M = r + 8*half, N = lrow ---
#pragma unroll
  for (int mt = 0; mt < 2; ++mt)
#pragma unroll
    for (int nt = 0; nt < 2; ++nt) {
      int n = n0 + wn * 32 + nt * 16 + lrow;
      if (!NGUARD || n < N) {
        int mbase = m0 + wm * 32 + mt * 16 + lhalf * 8;
#pragma unroll
        for (int r = 0; r < 8; ++r)
          C[(size_t)(mbase + r) * N + n] = acc[mt][nt][r];
      }
    }
}

// --------------------- depthwise causal conv + SiLU ------------------------
__global__ void dwconv_silu_kernel(const float* __restrict__ xz, const float* __restrict__ cw,
                                   const float* __restrict__ cb,
                                   float* __restrict__ xc, _Float16* __restrict__ xc16) {
  size_t idx = blockIdx.x * (size_t)blockDim.x + threadIdx.x;   // MT*DI
  if (idx >= MT * NDI) return;
  int d = (int)(idx % NDI);
  size_t i = idx / NDI;
  int t = (int)(i % TT);
  size_t b = i / TT;
  float acc = cb[d];
#pragma unroll
  for (int j = 0; j < NDC; ++j) {
    int tt = t - (NDC - 1) + j;
    if (tt >= 0) acc += xz[((b * TT + tt) * (2 * NDI)) + d] * cw[d * NDC + j];
  }
  float s = acc / (1.f + __expf(-acc));
  xc[idx] = s;
  xc16[idx] = (_Float16)s;
}

// ------------------------------ softplus -----------------------------------
__global__ void softplus_kernel(float* __restrict__ dt, const float* __restrict__ dtb) {
  size_t idx = blockIdx.x * (size_t)blockDim.x + threadIdx.x;   // MT*DI
  if (idx >= MT * NDI) return;
  int d = (int)(idx % NDI);
  float v = dt[idx] + dtb[d];
  dt[idx] = (v > 20.f) ? v : log1pf(__expf(v));
}

// --------------------------- selective scan --------------------------------
// grid = B * (DI/128), block = 128 threads. One thread owns channel d, 16 states.
__global__ __launch_bounds__(128) void scan_kernel(const float* __restrict__ dt,
                                                   const float* __restrict__ xc,
                                                   const float* __restrict__ dbl,
                                                   const float* __restrict__ A_log,
                                                   float* __restrict__ ys) {
  int bid = blockIdx.x;
  int b = bid >> 2;            // DI/128 == 4 chunks
  int dch = bid & 3;
  int d = dch * 128 + threadIdx.x;

  float Arow[NDS];
#pragma unroll
  for (int s = 0; s < NDS; ++s) Arow[s] = -__expf(A_log[d * NDS + s]);
  float hst[NDS];
#pragma unroll
  for (int s = 0; s < NDS; ++s) hst[s] = 0.f;

  __shared__ float sBC[2 * NDS];
  for (int t = 0; t < TT; ++t) {
    size_t i = (size_t)b * TT + t;
    if (threadIdx.x < 2 * NDS) sBC[threadIdx.x] = dbl[i * 48 + NDR + threadIdx.x];
    __syncthreads();
    float dtv = dt[i * NDI + d];
    float xv  = xc[i * NDI + d];
    float dtx = dtv * xv;
    float y = 0.f;
#pragma unroll
    for (int s = 0; s < NDS; ++s) {
      hst[s] = hst[s] * __expf(dtv * Arow[s]) + dtx * sBC[s];
      y += hst[s] * sBC[NDS + s];
    }
    ys[i * NDI + d] = y;
    __syncthreads();
  }
}

// ------------------- gate: (ys + D*xc) * silu(z) -> f16 --------------------
__global__ void gate_kernel(const float* __restrict__ ys, const float* __restrict__ xc,
                            const float* __restrict__ xz, const float* __restrict__ Dp,
                            _Float16* __restrict__ y16) {
  size_t idx = blockIdx.x * (size_t)blockDim.x + threadIdx.x;   // MT*DI
  if (idx >= MT * NDI) return;
  int d = (int)(idx % NDI);
  size_t i = idx / NDI;
  float z = xz[i * (2 * NDI) + NDI + d];
  float y = (ys[idx] + Dp[d] * xc[idx]) * (z / (1.f + __expf(-z)));
  y16[idx] = (_Float16)y;
}

// -------------------- residual add + LayerNorm (H=256) ---------------------
__global__ __launch_bounds__(256) void add_ln_kernel(float* __restrict__ h,
                                                     const float* __restrict__ lo,
                                                     const float* __restrict__ g,
                                                     const float* __restrict__ bta) {
  size_t i = blockIdx.x;              // row (B*T rows)
  int tid = threadIdx.x;              // 0..255
  float v = h[i * HH + tid] + lo[i * HH + tid];
  float s = v, s2 = v * v;
#pragma unroll
  for (int off = 16; off > 0; off >>= 1) {
    s  += __shfl_down(s,  off, 32);
    s2 += __shfl_down(s2, off, 32);
  }
  __shared__ float ss[8], ss2[8];
  int wid = tid >> 5, lid = tid & 31;
  if (lid == 0) { ss[wid] = s; ss2[wid] = s2; }
  __syncthreads();
  if (wid == 0) {
    s  = (lid < 8) ? ss[lid]  : 0.f;
    s2 = (lid < 8) ? ss2[lid] : 0.f;
#pragma unroll
    for (int off = 4; off > 0; off >>= 1) {
      s  += __shfl_down(s,  off, 32);
      s2 += __shfl_down(s2, off, 32);
    }
    if (lid == 0) { ss[0] = s; ss2[0] = s2; }
  }
  __syncthreads();
  float mean = ss[0] * (1.f / HH);
  float var  = ss2[0] * (1.f / HH) - mean * mean;
  h[i * HH + tid] = (v - mean) * rsqrtf(var + 1e-5f) * g[tid] + bta[tid];
}

// --------------------------- mean over T + head ----------------------------
__global__ void time_mean_kernel(const float* __restrict__ h, float* __restrict__ meanb) {
  size_t idx = blockIdx.x * (size_t)blockDim.x + threadIdx.x;   // B*H = 4096
  if (idx >= (size_t)BB * HH) return;
  int b = (int)(idx / HH), hc = (int)(idx % HH);
  float s = 0.f;
  for (int t = 0; t < TT; ++t) s += h[((size_t)b * TT + t) * HH + hc];
  meanb[idx] = s * (1.f / TT);
}

__global__ void head_kernel(const float* __restrict__ meanb, const float* __restrict__ ow,
                            const float* __restrict__ ob, float* __restrict__ out) {
  int idx = blockIdx.x * blockDim.x + threadIdx.x;              // B*OUT = 160
  if (idx >= BB * NOUT) return;
  int b = idx / NOUT, o = idx % NOUT;
  float s = ob[o];
  for (int hc = 0; hc < HH; ++hc) s += meanb[b * HH + hc] * ow[o * HH + hc];
  out[idx] = s;
}

// ---------------------------------------------------------------------------
extern "C" void kernel_launch(void* const* d_in, const int* in_sizes, int n_in,
                              void* d_out, int out_size, void* d_ws, size_t ws_size,
                              hipStream_t stream) {
  (void)in_sizes; (void)n_in; (void)out_size; (void)ws_size;
  const float* x    = (const float*)d_in[0];
  const float* w3   = (const float*)d_in[1];
  const float* b3   = (const float*)d_in[2];
  const float* w5   = (const float*)d_in[3];
  const float* b5   = (const float*)d_in[4];
  const float* w7   = (const float*)d_in[5];
  const float* b7   = (const float*)d_in[6];
  const float* ipw  = (const float*)d_in[7];   // (5, 1024, 256)
  const float* cw   = (const float*)d_in[8];   // (5, 512, 4)
  const float* cb   = (const float*)d_in[9];   // (5, 512)
  const float* xpw  = (const float*)d_in[10];  // (5, 48, 512)
  const float* dtw  = (const float*)d_in[11];  // (5, 512, 16)
  const float* dtb  = (const float*)d_in[12];  // (5, 512)
  const float* Alog = (const float*)d_in[13];  // (5, 512, 16)
  const float* Dp   = (const float*)d_in[14];  // (5, 512)
  const float* opw  = (const float*)d_in[15];  // (5, 256, 512)
  const float* ln_g = (const float*)d_in[16];
  const float* ln_b = (const float*)d_in[17];
  const float* ow   = (const float*)d_in[18];
  const float* ob   = (const float*)d_in[19];
  float* out = (float*)d_out;

  // workspace carve-up
  char* p = (char*)d_ws;
  float* hbuf  = (float*)p;  p += MT * HH      * sizeof(float);    // 32 MB
  float* xz    = (float*)p;  p += MT * 2*NDI   * sizeof(float);    // 128 MB
  float* xc    = (float*)p;  p += MT * NDI     * sizeof(float);    // 64 MB
  float* dtbuf = (float*)p;  p += MT * NDI     * sizeof(float);    // 64 MB (dt linear, softplus in place)
  float* dbl   = (float*)p;  p += MT * 48      * sizeof(float);    // 6 MB
  float* ysb   = (float*)p;  p += MT * NDI     * sizeof(float);    // 64 MB
  float* lo    = (float*)p;  p += MT * HH      * sizeof(float);    // 32 MB
  float* meanb = (float*)p;  p += (size_t)BB * HH * sizeof(float);
  _Float16* h16  = (_Float16*)p; p += MT * HH  * sizeof(_Float16); // 16 MB
  _Float16* xc16 = (_Float16*)p; p += MT * NDI * sizeof(_Float16); // 32 MB
  _Float16* y16  = (_Float16*)p; p += MT * NDI * sizeof(_Float16); // 32 MB
  _Float16* dtA  = (_Float16*)p; p += MT * 32  * sizeof(_Float16); // 2 MB
  _Float16* wB   = (_Float16*)p; p += (size_t)HH * 2 * NDI * sizeof(_Float16); // 512 KB, reused

  const int BS = 256;
  auto blocks = [](size_t n, int bs) { return (unsigned)((n + bs - 1) / bs); };

  // front-end conv stack -> h
  front_conv_kernel<<<blocks(MT * HH, BS), BS, 0, stream>>>(x, w3, b3, w5, b5, w7, b7, hbuf);

  for (int L = 0; L < NDEPTH; ++L) {
    // ---- in_proj: xz = h * ipw^T  (M=32768, N=1024, K=256) ----
    cvt_f16_kernel<<<blocks(MT * HH, BS), BS, 0, stream>>>(hbuf, h16, MT * HH);
    cvt_transpose_pad_kernel<<<blocks((size_t)HH * 2 * NDI, BS), BS, 0, stream>>>(
        ipw + (size_t)L * 2 * NDI * HH, wB, 2 * NDI, HH, HH);
    wmma_gemm_kernel<false><<<dim3((unsigned)(MT / 128), (2 * NDI) / 64), BS, 0, stream>>>(
        h16, wB, xz, (int)MT, 2 * NDI, HH);

    // ---- depthwise causal conv + SiLU -> xc (f32 + f16) ----
    dwconv_silu_kernel<<<blocks(MT * NDI, BS), BS, 0, stream>>>(
        xz, cw + (size_t)L * NDI * NDC, cb + (size_t)L * NDI, xc, xc16);

    // ---- x_proj: dbl = xc * xpw^T  (N=48 -> guarded instance, K=512) ----
    cvt_transpose_pad_kernel<<<blocks((size_t)NDI * 48, BS), BS, 0, stream>>>(
        xpw + (size_t)L * 48 * NDI, wB, 48, NDI, NDI);
    wmma_gemm_kernel<true><<<dim3((unsigned)(MT / 128), 1), BS, 0, stream>>>(
        xc16, wB, dbl, (int)MT, 48, NDI);

    // ---- dt_proj: dt = softplus(dtr * dtw^T + b)  (K=16 padded to 32, N=512) ----
    pack_dtr_kernel<<<blocks(MT * 32, BS), BS, 0, stream>>>(dbl, dtA);
    cvt_transpose_pad_kernel<<<blocks((size_t)32 * NDI, BS), BS, 0, stream>>>(
        dtw + (size_t)L * NDI * NDR, wB, NDI, NDR, 32);
    wmma_gemm_kernel<false><<<dim3((unsigned)(MT / 128), NDI / 64), BS, 0, stream>>>(
        dtA, wB, dtbuf, (int)MT, NDI, 32);
    softplus_kernel<<<blocks(MT * NDI, BS), BS, 0, stream>>>(dtbuf, dtb + (size_t)L * NDI);

    // ---- selective scan ----
    scan_kernel<<<BB * (NDI / 128), 128, 0, stream>>>(
        dtbuf, xc, dbl, Alog + (size_t)L * NDI * NDS, ysb);

    // ---- gate -> y16 ----
    gate_kernel<<<blocks(MT * NDI, BS), BS, 0, stream>>>(ysb, xc, xz, Dp + (size_t)L * NDI, y16);

    // ---- out_proj: lo = y * opw^T  (N=256, K=512) ----
    cvt_transpose_pad_kernel<<<blocks((size_t)NDI * HH, BS), BS, 0, stream>>>(
        opw + (size_t)L * HH * NDI, wB, HH, NDI, NDI);
    wmma_gemm_kernel<false><<<dim3((unsigned)(MT / 128), HH / 64), BS, 0, stream>>>(
        y16, wB, lo, (int)MT, HH, NDI);

    // ---- residual + LayerNorm ----
    add_ln_kernel<<<(unsigned)MT, HH, 0, stream>>>(hbuf, lo, ln_g, ln_b);
  }

  // mean over T, then linear head
  time_mean_kernel<<<blocks((size_t)BB * HH, BS), BS, 0, stream>>>(hbuf, meanb);
  head_kernel<<<1, BS, 0, stream>>>(meanb, ow, ob, out);
}
